// Heads_20169166422419
// MI455X (gfx1250) — matrix operand order repaired and checked by reading, not compiled, not run on hardware
//
#include <hip/hip_runtime.h>
#include <hip/hip_bf16.h>
#include <math.h>

// ---------------------------------------------------------------------------
// Multi-head attention forward on MI455X (gfx1250).
// All GEMMs: v_wmma_f32_16x16x32_bf16, 128x128 block tiles double-buffered in
// LDS via global_load_async_to_lds_b128 (ASYNCcnt) -> ds_load fragments.
//
// B=16, T=512, E=4096, H=8, HD=512
//   1) f32->bf16 convert of data; convert+transpose of Wq/Wk/Wv/Wo
//   2) QKV projections -> Q,K ([H,BT,HD]) and V transposed ([H,HD,BT])
//   3) scores = Q K^T / sqrt(E) + causal mask (f32)
//   4) row softmax -> P (bf16)
//   5) Ctx = P @ V into concatenated [B,T,E] bf16
//   6) out = Ctx @ Wo + bo (f32)
// ---------------------------------------------------------------------------

constexpr int BB = 16;
constexpr int TT = 512;
constexpr int EE = 4096;
constexpr int HH = 8;
constexpr int HD = 512;
constexpr long BT = (long)BB * TT; // 8192

typedef __attribute__((ext_vector_type(16))) __bf16 bf16x16;
typedef __attribute__((ext_vector_type(8)))  __bf16 bf16x8;
typedef __attribute__((ext_vector_type(8)))  float  f32x8;

__device__ __forceinline__ bf16x16 cat8(bf16x8 lo, bf16x8 hi) {
  return __builtin_shufflevector(lo, hi, 0,1,2,3,4,5,6,7,8,9,10,11,12,13,14,15);
}

// ---- async global->LDS copy (CDNA5 ASYNCcnt path) --------------------------
// Each lane copies 16B. LDS destination address = low 32 bits of the generic
// pointer to __shared__ (aperture form: [31:0] is the LDS byte offset).
__device__ __forceinline__ void async_copy_b128(const __bf16* g, const __bf16* l) {
  unsigned loff = (unsigned)(unsigned long long)l;
  asm volatile("global_load_async_to_lds_b128 %0, %1, off"
               :: "v"(loff), "v"(g) : "memory");
}
__device__ __forceinline__ void wait_async0() {
  asm volatile("s_wait_asynccnt 0" ::: "memory");
}

// Stage a 128-row x 32-half (8KB) tile: 256 threads x 2 x 16B.
__device__ __forceinline__ void stage_tile(const __bf16* __restrict__ g, long ld,
                                           __bf16* lbuf) {
  const int t = threadIdx.x;
  const int c = (t & 3) << 3;   // half offset within row (0,8,16,24)
  const int r = t >> 2;         // 0..63
  async_copy_b128(g + (long)r * ld + c,        lbuf + r * 32 + c);
  async_copy_b128(g + (long)(r + 64) * ld + c, lbuf + (r + 64) * 32 + c);
}

// ---- LDS fragment loads (ISA 7.12.2 layouts), tile stored [128][32] halves -
// A frag 16x32: lane L -> row (L&15); K chunks (L>>4)*8 + {0..7} and {16..23}
__device__ __forceinline__ bf16x16 lds_afrag(const __bf16* tile, int row_base) {
  const int lane = threadIdx.x & 31;
  const __bf16* p = tile + (row_base + (lane & 15)) * 32 + ((lane >> 4) << 3);
  bf16x8 lo = *(const bf16x8*)p;
  bf16x8 hi = *(const bf16x8*)(p + 16);
  return cat8(lo, hi);
}
// B frag 32x16 from Bt rows: lane L -> col (L&15); K chunks (L>>4)*8 + {0..7} and {16..23}
__device__ __forceinline__ bf16x16 lds_bfrag(const __bf16* tile, int col_base) {
  const int lane = threadIdx.x & 31;
  const __bf16* p = tile + (col_base + (lane & 15)) * 32 + ((lane >> 4) << 3);
  bf16x8 lo = *(const bf16x8*)p;
  bf16x8 hi = *(const bf16x8*)(p + 16);
  return cat8(lo, hi);
}

struct AccW { f32x8 c[4][2]; };   // one wave: 64(M) x 32(N)

// 128x128 block tile GEMM, K-step 32, LDS double buffered.
// 256 threads = 8 waves arranged 2(M:64) x 4(N:32).
__device__ __forceinline__ void block_gemm(const __bf16* Ablk, long lda,
                                           const __bf16* Btblk, long ldb,
                                           int K, AccW& acc) {
  __shared__ __bf16 Abuf[2][128 * 32];
  __shared__ __bf16 Bbuf[2][128 * 32];
  const int w    = threadIdx.x >> 5;
  const int wrow = (w & 1) << 6;   // 0 / 64
  const int wcol = (w >> 1) << 5;  // 0 / 32 / 64 / 96
  const int nk   = K >> 5;

  stage_tile(Ablk,  lda, Abuf[0]);
  stage_tile(Btblk, ldb, Bbuf[0]);
  wait_async0();
  __syncthreads();

  for (int kt = 0; kt < nk; ++kt) {
    const int cur = kt & 1;
    if (kt + 1 < nk) {                    // prefetch next K-slice into other buf
      stage_tile(Ablk  + (long)(kt + 1) * 32, lda, Abuf[cur ^ 1]);
      stage_tile(Btblk + (long)(kt + 1) * 32, ldb, Bbuf[cur ^ 1]);
    }
    const __bf16* At = Abuf[cur];
    const __bf16* Bt = Bbuf[cur];
    bf16x16 b0 = lds_bfrag(Bt, wcol);
    bf16x16 b1 = lds_bfrag(Bt, wcol + 16);
    for (int mi = 0; mi < 4; ++mi) {
      bf16x16 a = lds_afrag(At, wrow + (mi << 4));
      acc.c[mi][0] = __builtin_amdgcn_wmma_f32_16x16x32_bf16(false, a, false, b0, (short)0, acc.c[mi][0], false, false);
      acc.c[mi][1] = __builtin_amdgcn_wmma_f32_16x16x32_bf16(false, a, false, b1, (short)0, acc.c[mi][1], false, false);
    }
    wait_async0();
    __syncthreads();
  }
}

__device__ __forceinline__ void tile_origin(int& row0, int& col0) {
  const int w = threadIdx.x >> 5;
  row0 = blockIdx.y * 128 + ((w & 1) << 6);
  col0 = blockIdx.x * 128 + ((w >> 1) << 5);
}

// C/D layout: lane L -> col (L&15), VGPR v -> row (L>>4)*8 + v
#define EPILOGUE_LOOP(body)                                   \
  {                                                           \
    const int lane  = threadIdx.x & 31;                       \
    const int nloc  = lane & 15;                              \
    const int mbase = (lane >> 4) << 3;                       \
    for (int mi = 0; mi < 4; ++mi)                            \
      for (int nj = 0; nj < 2; ++nj)                          \
        for (int v = 0; v < 8; ++v) {                         \
          const int m = row0 + (mi << 4) + mbase + v;         \
          const int n = col0 + (nj << 4) + nloc;              \
          float val = acc.c[mi][nj][v];                       \
          body;                                               \
        }                                                     \
  }

// ---- conversion kernels ----------------------------------------------------
__global__ void cvt_f32_bf16(const float* __restrict__ in, __bf16* __restrict__ out, long n) {
  long i = (long)blockIdx.x * blockDim.x + threadIdx.x;
  if (i < n) out[i] = (__bf16)in[i];
}

// in: [z][rows, cols] f32  -> out: [z][cols, rows] bf16
__global__ void transpose_cvt(const float* __restrict__ in, __bf16* __restrict__ out,
                              int rows, int cols) {
  long base = (long)blockIdx.z * rows * cols;
  long i = (long)blockIdx.x * blockDim.x + threadIdx.x;
  if (i < (long)rows * cols) {
    int r = (int)(i / cols), c = (int)(i % cols);
    out[base + (long)c * rows + r] = (__bf16)in[base + i];
  }
}

// ---- GEMM kernels ----------------------------------------------------------

// Out[h] = X @ W[h]; Wt pre-transposed [H][N,K]. transposed=1 -> Out [H][N,M].
__global__ __launch_bounds__(256)
void proj_kernel(const __bf16* __restrict__ X, const __bf16* __restrict__ Wt,
                 __bf16* __restrict__ Out, int Mtotal, int Kdim, int Ndim, int transposed) {
  int row0, col0; tile_origin(row0, col0);
  const __bf16* Wth = Wt + (long)blockIdx.z * Ndim * Kdim;
  AccW acc = {};
  block_gemm(X + (long)blockIdx.y * 128 * Kdim, Kdim,
             Wth + (long)blockIdx.x * 128 * Kdim, Kdim, Kdim, acc);
  __bf16* Oz = Out + (long)blockIdx.z * Mtotal * Ndim;
  if (!transposed) {
    EPILOGUE_LOOP(Oz[(long)m * Ndim + n] = (__bf16)val;)
  } else {
    EPILOGUE_LOOP(Oz[(long)n * Mtotal + m] = (__bf16)val;)
  }
}

// S[z] = (Q[z] @ K[z]^T) * scale with causal mask; z = h*B + b
__global__ __launch_bounds__(256)
void scores_kernel(const __bf16* __restrict__ Q, const __bf16* __restrict__ Kb,
                   float* __restrict__ S, float scale) {
  int row0, col0; tile_origin(row0, col0);
  const long zoff = (long)blockIdx.z * TT * HD;
  AccW acc = {};
  block_gemm(Q + zoff + (long)blockIdx.y * 128 * HD, HD,
             Kb + zoff + (long)blockIdx.x * 128 * HD, HD, HD, acc);
  float* Sz = S + (long)blockIdx.z * TT * TT;
  EPILOGUE_LOOP({
    val *= scale;
    if (n > m) val = -INFINITY;          // causal mask
    Sz[(long)m * TT + n] = val;
  })
}

// row softmax: one 256-thread block per row of length TT
__global__ __launch_bounds__(256)
void softmax_rows(const float* __restrict__ S, __bf16* __restrict__ P) {
  __shared__ float red[256];
  const long row = blockIdx.x;
  const float* s = S + row * TT;
  __bf16* p = P + row * TT;
  const int tid = threadIdx.x;
  float m = -INFINITY;
  for (int i = tid; i < TT; i += 256) m = fmaxf(m, s[i]);
  red[tid] = m; __syncthreads();
  for (int st = 128; st > 0; st >>= 1) { if (tid < st) red[tid] = fmaxf(red[tid], red[tid + st]); __syncthreads(); }
  m = red[0]; __syncthreads();
  float sum = 0.f;
  for (int i = tid; i < TT; i += 256) sum += __expf(s[i] - m);
  red[tid] = sum; __syncthreads();
  for (int st = 128; st > 0; st >>= 1) { if (tid < st) red[tid] += red[tid + st]; __syncthreads(); }
  const float inv = 1.0f / red[0];
  for (int i = tid; i < TT; i += 256) p[i] = (__bf16)(__expf(s[i] - m) * inv);
}

// Ctx[b*T+t, h*HD+d] = sum_s P[z][t,s] V[z][s,d]; Vt is [H][HD][BT], z = h*B+b
__global__ __launch_bounds__(256)
void pv_kernel(const __bf16* __restrict__ P, const __bf16* __restrict__ Vt,
               __bf16* __restrict__ Ctx) {
  int row0, col0; tile_origin(row0, col0);
  const int z = blockIdx.z, h = z / BB, b = z % BB;
  const __bf16* A  = P + (long)z * TT * TT + (long)blockIdx.y * 128 * TT;
  const __bf16* Bt = Vt + (long)h * HD * BT + (long)b * TT + (long)blockIdx.x * 128 * BT;
  AccW acc = {};
  block_gemm(A, TT, Bt, BT, TT, acc);
  EPILOGUE_LOOP(Ctx[((long)b * TT + m) * EE + (long)h * HD + n] = (__bf16)val;)
}

// out = Ctx @ Wo + bo (WtO is [E,E] transposed), f32 output
__global__ __launch_bounds__(256)
void outproj_kernel(const __bf16* __restrict__ Ctx, const __bf16* __restrict__ WtO,
                    const float* __restrict__ bo, float* __restrict__ Y) {
  int row0, col0; tile_origin(row0, col0);
  AccW acc = {};
  block_gemm(Ctx + (long)blockIdx.y * 128 * EE, EE,
             WtO + (long)blockIdx.x * 128 * EE, EE, EE, acc);
  EPILOGUE_LOOP(Y[(long)m * EE + n] = val + bo[n];)
}

// ---------------------------------------------------------------------------
extern "C" void kernel_launch(void* const* d_in, const int* in_sizes, int n_in,
                              void* d_out, int out_size, void* d_ws, size_t ws_size,
                              hipStream_t stream) {
  (void)in_sizes; (void)n_in; (void)out_size; (void)ws_size;
  const float* data = (const float*)d_in[0];
  const float* Wq   = (const float*)d_in[1];
  const float* Wk   = (const float*)d_in[2];
  const float* Wv   = (const float*)d_in[3];
  const float* Wo   = (const float*)d_in[4];
  const float* bo   = (const float*)d_in[5];
  float* out = (float*)d_out;
  char*  ws  = (char*)d_ws;

  const size_t szXbf = (size_t)BT * EE * 2;          // 67.1 MB
  const size_t szW   = (size_t)HH * HD * EE * 2;     // 33.5 MB each
  const size_t szWo  = (size_t)EE * EE * 2;          // 33.5 MB
  const size_t szQKV = (size_t)HH * BT * HD * 2;     // 67.1 MB each

  __bf16* Xbf = (__bf16*)(ws);
  __bf16* WtQ = (__bf16*)(ws + szXbf);
  __bf16* WtK = (__bf16*)(ws + szXbf + szW);
  __bf16* WtV = (__bf16*)(ws + szXbf + 2 * szW);
  __bf16* WtO = (__bf16*)(ws + szXbf + 3 * szW);
  __bf16* Qb  = (__bf16*)(ws + szXbf + 3 * szW + szWo);
  __bf16* Kb  = (__bf16*)((char*)Qb + szQKV);
  __bf16* Vt  = (__bf16*)((char*)Kb + szQKV);
  // Aliased regions (dead-after-use reuse):
  float*  S   = (float*)ws;   // over Xbf+WtQ+WtK exactly (ends at WtV); free after projections
  __bf16* P   = Qb;           // Q dead after scores
  __bf16* Ctx = Kb;           // K dead after scores

  dim3 blk(256);

  // 1) conversions / transposes
  cvt_f32_bf16<<<(unsigned)((BT * EE + 255) / 256), blk, 0, stream>>>(data, Xbf, BT * EE);
  transpose_cvt<<<dim3((unsigned)(((long)EE * HD + 255) / 256), 1, HH), blk, 0, stream>>>(Wq, WtQ, EE, HD);
  transpose_cvt<<<dim3((unsigned)(((long)EE * HD + 255) / 256), 1, HH), blk, 0, stream>>>(Wk, WtK, EE, HD);
  transpose_cvt<<<dim3((unsigned)(((long)EE * HD + 255) / 256), 1, HH), blk, 0, stream>>>(Wv, WtV, EE, HD);
  transpose_cvt<<<dim3((unsigned)(((long)EE * EE + 255) / 256), 1, 1), blk, 0, stream>>>(Wo, WtO, EE, EE);

  // 2) QKV projections: M=BT, N=HD, K=E, batched over heads
  dim3 gproj(HD / 128, (unsigned)(BT / 128), HH);
  proj_kernel<<<gproj, blk, 0, stream>>>(Xbf, WtQ, Qb, (int)BT, EE, HD, 0);
  proj_kernel<<<gproj, blk, 0, stream>>>(Xbf, WtK, Kb, (int)BT, EE, HD, 0);
  proj_kernel<<<gproj, blk, 0, stream>>>(Xbf, WtV, Vt, (int)BT, EE, HD, 1);

  // 3) scores (scale = 1/sqrt(E) = 1/64)
  dim3 gsc(TT / 128, TT / 128, BB * HH);
  scores_kernel<<<gsc, blk, 0, stream>>>(Qb, Kb, S, 1.0f / 64.0f);

  // 4) softmax over rows
  softmax_rows<<<(unsigned)(BB * HH * TT), blk, 0, stream>>>(S, P);

  // 5) P @ V -> concatenated context
  dim3 gpv(HD / 128, TT / 128, BB * HH);
  pv_kernel<<<gpv, blk, 0, stream>>>(P, Vt, Ctx);

  // 6) out projection + bias
  dim3 gout(EE / 128, (unsigned)(BT / 128), 1);
  outproj_kernel<<<gout, blk, 0, stream>>>(Ctx, WtO, bo, out);
}
